// MyCA_59691455480021
// MI455X (gfx1250) — matrix operand order repaired and checked by reading, not compiled, not run on hardware
//
#include <hip/hip_runtime.h>
#include <hip/hip_bf16.h>
#include <stdint.h>

#define NB    16       // batch
#define CIN   512
#define CR    128
#define HW    9216     // 96*96
#define KC    32       // WMMA K per chunk (bf16)
#define LDS_ST 40      // LDS row stride in ushorts (80 bytes, 16B aligned, bank-spread)

typedef __attribute__((ext_vector_type(16))) __bf16 v16bf;
typedef __attribute__((ext_vector_type(2)))  __bf16 v2bf;
typedef __attribute__((ext_vector_type(8)))  float  v8f;

union FragU { v16bf v; uint4 q[2]; };

struct R4f4 { float4 v[4]; };   // fp32 staging regs (4 x float4)
struct R2u4 { uint4  v[2]; };   // bf16 staging regs (2 x uint4)

// native fptrunc -> RNE bf16; compiler selects v_cvt_*bf16_f32 forms
__device__ __forceinline__ unsigned short f32_to_bf16(float f) {
    return __builtin_bit_cast(unsigned short, (__bf16)f);
}
__device__ __forceinline__ unsigned pack_bf16(float lo, float hi) {
    v2bf p;
    p.x = (__bf16)lo;
    p.y = (__bf16)hi;
    return __builtin_bit_cast(unsigned, p);
}

// ---------------- fragment loads from LDS (ISA 16-bit A/B layouts) -------------
// A tile in LDS: [128 rows (M)][KC], row stride LDS_ST ushorts.
// A frag 16x32: lanes 0-15 row m0+l, K {0..7}U{16..23}; lanes 16-31 K {8..15}U{24..31}
__device__ __forceinline__ v16bf load_fragA(const unsigned short* lds, int m0, int lane) {
    int half = lane >> 4, lr = lane & 15;
    const unsigned short* row = lds + (m0 + lr) * LDS_ST + 8 * half;
    FragU f;
    f.q[0] = *(const uint4*)(row);         // K kb..kb+7
    f.q[1] = *(const uint4*)(row + 16);    // K kb+16..kb+23
    return f.v;
}
// B tile in LDS: [128 rows (N)][KC] (column-of-B contiguous along K).
// B frag 32x16: lanes 0-15 col n0+l, K {0..7}U{16..23}; lanes 16-31 K {8..15}U{24..31}
__device__ __forceinline__ v16bf load_fragB(const unsigned short* lds, int n0, int lane) {
    int half = lane >> 4, lr = lane & 15;
    const unsigned short* col = lds + (n0 + lr) * LDS_ST + 8 * half;
    FragU f;
    f.q[0] = *(const uint4*)(col);
    f.q[1] = *(const uint4*)(col + 16);
    return f.v;
}

// ------------- split staging: global->regs, then regs->LDS (ping-pong) ---------
// A tile [128 rows][KC] from fp32 row-major src (ld floats), cols c0..c0+31
__device__ __forceinline__ R4f4 loadA_f32(const float* src, int ld, int c0) {
    int t = threadIdx.x; R4f4 r;
#pragma unroll
    for (int i = 0; i < 4; ++i) {
        int fi = t + 256 * i;              // float4 idx 0..1023
        int row = fi >> 3, k4 = fi & 7;    // 8 float4 per row
        r.v[i] = *(const float4*)(src + (size_t)row * ld + c0 + k4 * 4);
    }
    return r;
}
__device__ __forceinline__ void storeA_f32(unsigned short* lds, const R4f4& r) {
    int t = threadIdx.x;
#pragma unroll
    for (int i = 0; i < 4; ++i) {
        int fi = t + 256 * i;
        int row = fi >> 3, k4 = fi & 7;
        uint2 p;
        p.x = pack_bf16(r.v[i].x, r.v[i].y);
        p.y = pack_bf16(r.v[i].z, r.v[i].w);
        *(uint2*)(lds + row * LDS_ST + k4 * 4) = p;
    }
}
// A tile [128 rows][KC] straight copy from bf16 row-major src (ld ushorts)
__device__ __forceinline__ R2u4 loadA_b16(const unsigned short* src, int ld, int k0) {
    int t = threadIdx.x; R2u4 r;
#pragma unroll
    for (int i = 0; i < 2; ++i) {
        int fi = t + 256 * i;              // uint4 idx 0..511
        int row = fi >> 2, k8 = fi & 3;    // 4 uint4 per row
        r.v[i] = *(const uint4*)(src + (size_t)row * ld + k0 + k8 * 8);
    }
    return r;
}
__device__ __forceinline__ void storeA_b16(unsigned short* lds, const R2u4& r) {
    int t = threadIdx.x;
#pragma unroll
    for (int i = 0; i < 2; ++i) {
        int fi = t + 256 * i;
        int row = fi >> 2, k8 = fi & 3;
        *(uint4*)(lds + row * LDS_ST + k8 * 8) = r.v[i];
    }
}
// B transpose from fp32 src [C][HW]: each thread takes 4 consecutive k-rows x 4 n,
// packs bf16 (k,k+1) pairs -> b64 LDS stores (4x fewer DS ops than b16 scatter)
__device__ __forceinline__ R4f4 loadB_Tf32(const float* src, int ld, int c0, int nb) {
    int t = threadIdx.x; R4f4 r;
    int kp = t >> 5, n4 = t & 31;          // kp: 8 groups of 4 k-rows; n4: 32 col-groups
#pragma unroll
    for (int rr = 0; rr < 4; ++rr)
        r.v[rr] = *(const float4*)(src + (size_t)(c0 + kp * 4 + rr) * ld + nb + n4 * 4);
    return r;
}
__device__ __forceinline__ void storeB_Tf32(unsigned short* lds, const R4f4& r) {
    int t = threadIdx.x;
    int kp = t >> 5, n4 = t & 31;
    const float* f = (const float*)&r.v[0];   // f[rr*4 + j]
#pragma unroll
    for (int j = 0; j < 4; ++j) {
        uint2 p;
        p.x = pack_bf16(f[0 * 4 + j], f[1 * 4 + j]);   // k = 4kp, 4kp+1
        p.y = pack_bf16(f[2 * 4 + j], f[3 * 4 + j]);   // k = 4kp+2, 4kp+3
        *(uint2*)(lds + (n4 * 4 + j) * LDS_ST + kp * 4) = p;
    }
}
// B transpose from bf16 src [K][N]: 2 consecutive k-rows x 8 n per thread -> b32 stores
__device__ __forceinline__ R2u4 loadB_Tb16(const unsigned short* src, int ld, int k0, int nb) {
    int t = threadIdx.x; R2u4 r;
    int kp = t >> 4, n8 = t & 15;          // kp: 16 pairs of k-rows; n8: 16 col-groups
#pragma unroll
    for (int rr = 0; rr < 2; ++rr)
        r.v[rr] = *(const uint4*)(src + (size_t)(k0 + kp * 2 + rr) * ld + nb + n8 * 8);
    return r;
}
__device__ __forceinline__ void storeB_Tb16(unsigned short* lds, const R2u4& r) {
    int t = threadIdx.x;
    int kp = t >> 4, n8 = t & 15;
    const unsigned short* s0 = (const unsigned short*)&r.v[0];
    const unsigned short* s1 = (const unsigned short*)&r.v[1];
#pragma unroll
    for (int j = 0; j < 8; ++j) {
        unsigned p = (unsigned)s0[j] | ((unsigned)s1[j] << 16);  // k = 2kp, 2kp+1
        *(unsigned*)(lds + (n8 * 8 + j) * LDS_ST + kp * 2) = p;
    }
}

// 8 WMMA per wave per K-chunk: 2 M-tiles x 4 N-tiles of 16x16
__device__ __forceinline__ void wmma_chunk(const unsigned short* lA, const unsigned short* lB,
                                           int wm, int wn, int lane, v8f acc[2][4]) {
    v16bf a0 = load_fragA(lA, wm, lane);
    v16bf a1 = load_fragA(lA, wm + 16, lane);
#pragma unroll
    for (int j = 0; j < 4; ++j) {
        v16bf bj = load_fragB(lB, wn + 16 * j, lane);
        acc[0][j] = __builtin_amdgcn_wmma_f32_16x16x32_bf16(false, a0, false, bj,
                                                            (short)0, acc[0][j], false, false);
        acc[1][j] = __builtin_amdgcn_wmma_f32_16x16x32_bf16(false, a1, false, bj,
                                                            (short)0, acc[1][j], false, false);
    }
}

// =============== kernel 1: q/k/v projection + BN + PReLU -> bf16 plane =========
__global__ __launch_bounds__(256) void qkv_proj_kernel(
    const float* __restrict__ x, const float* __restrict__ Wt,
    const float* __restrict__ bias, const float* __restrict__ g,
    const float* __restrict__ beta, const float* __restrict__ mu,
    const float* __restrict__ var, const float* __restrict__ slope,
    unsigned short* __restrict__ dst, int has_bn)
{
    __shared__ alignas(16) unsigned short lA[2][128 * LDS_ST];
    __shared__ alignas(16) unsigned short lB[2][128 * LDS_ST];
    int b  = blockIdx.y;
    int nb = blockIdx.x * 128;
    const float* xb = x + (size_t)b * CIN * HW;
    unsigned short* db = dst + (size_t)b * CR * HW;

    int lane = threadIdx.x & 31, wave = threadIdx.x >> 5;
    int wm = (wave & 3) * 32, wn = (wave >> 2) * 64;
    v8f acc[2][4] = {};

    R4f4 ra = loadA_f32(Wt, CIN, 0);
    R4f4 rb = loadB_Tf32(xb, HW, 0, nb);
    storeA_f32(lA[0], ra);
    storeB_Tf32(lB[0], rb);
    __syncthreads();

    const int NCH = CIN / KC;                  // 16
    for (int c = 0; c < NCH; ++c) {
        int cur = c & 1;
        if (c + 1 < NCH) {
            if (c + 2 < NCH)
                __builtin_prefetch(xb + (size_t)(c + 2) * KC * HW + nb, 0, 0); // global_prefetch_b8
            ra = loadA_f32(Wt, CIN, (c + 1) * KC);       // global loads in flight...
            rb = loadB_Tf32(xb, HW, (c + 1) * KC, nb);
        }
        wmma_chunk(lA[cur], lB[cur], wm, wn, lane, acc); // ...while WMMA runs
        __syncthreads();
        if (c + 1 < NCH) {
            storeA_f32(lA[cur ^ 1], ra);
            storeB_Tf32(lB[cur ^ 1], rb);
        }
        __syncthreads();
    }

    float a = slope[0];
    int half = lane >> 4, lc = lane & 15;
#pragma unroll
    for (int i = 0; i < 2; ++i)
#pragma unroll
        for (int j = 0; j < 4; ++j)
#pragma unroll
            for (int r = 0; r < 8; ++r) {
                int m = wm + 16 * i + r + 8 * half;
                int n = nb + wn + 16 * j + lc;
                float v = acc[i][j][r] + bias[m];
                if (has_bn) {
                    float inv = g[m] * rsqrtf(var[m] + 1e-5f);
                    v = v * inv + (beta[m] - mu[m] * inv);
                }
                v = v >= 0.f ? v : a * v;
                db[(size_t)m * HW + n] = f32_to_bf16(v);
            }
}

// =============== kernel 2: alpha[b] += q . k^T over HW split ===================
__global__ __launch_bounds__(256) void alpha_kernel(
    const unsigned short* __restrict__ q, const unsigned short* __restrict__ k,
    float* __restrict__ alpha)
{
    __shared__ alignas(16) unsigned short lA[2][128 * LDS_ST];
    __shared__ alignas(16) unsigned short lB[2][128 * LDS_ST];
    int b = blockIdx.y;
    int kbase = blockIdx.x * 256;              // 36 splits * 256 = 9216
    const unsigned short* qb = q + (size_t)b * CR * HW;
    const unsigned short* kb = k + (size_t)b * CR * HW;

    int lane = threadIdx.x & 31, wave = threadIdx.x >> 5;
    int wm = (wave & 3) * 32, wn = (wave >> 2) * 64;
    v8f acc[2][4] = {};

    R2u4 ra = loadA_b16(qb, HW, kbase);
    R2u4 rk = loadA_b16(kb, HW, kbase);        // both operands row-major along K(=HW)
    storeA_b16(lA[0], ra);
    storeA_b16(lB[0], rk);
    __syncthreads();

    const int NCH = 256 / KC;                  // 8
    for (int c = 0; c < NCH; ++c) {
        int cur = c & 1;
        if (c + 1 < NCH) {
            ra = loadA_b16(qb, HW, kbase + (c + 1) * KC);
            rk = loadA_b16(kb, HW, kbase + (c + 1) * KC);
        }
        wmma_chunk(lA[cur], lB[cur], wm, wn, lane, acc);
        __syncthreads();
        if (c + 1 < NCH) {
            storeA_b16(lA[cur ^ 1], ra);
            storeA_b16(lB[cur ^ 1], rk);
        }
        __syncthreads();
    }

    float* ab = alpha + (size_t)b * CR * CR;
    int half = lane >> 4, lc = lane & 15;
#pragma unroll
    for (int i = 0; i < 2; ++i)
#pragma unroll
        for (int j = 0; j < 4; ++j)
#pragma unroll
            for (int r = 0; r < 8; ++r) {
                int m = wm + 16 * i + r + 8 * half;
                int n = wn + 16 * j + lc;
                atomicAdd(&ab[m * CR + n], acc[i][j][r]);   // global_atomic_add_f32
            }
}

// =============== kernel 3: row softmax 128-wide -> bf16 attn ===================
__global__ __launch_bounds__(128) void softmax_kernel(
    const float* __restrict__ alpha, unsigned short* __restrict__ attn)
{
    __shared__ float red[128];
    int row = blockIdx.x;                      // 0 .. NB*CR-1
    int t = threadIdx.x;
    float v = alpha[(size_t)row * CR + t];
    red[t] = v; __syncthreads();
    for (int s = 64; s > 0; s >>= 1) { if (t < s) red[t] = fmaxf(red[t], red[t + s]); __syncthreads(); }
    float mx = red[0]; __syncthreads();
    float e = __expf(v - mx);
    red[t] = e; __syncthreads();
    for (int s = 64; s > 0; s >>= 1) { if (t < s) red[t] += red[t + s]; __syncthreads(); }
    float sum = red[0];
    attn[(size_t)row * CR + t] = f32_to_bf16(e / sum);
}

// =============== kernel 4: y = attn . v  -> bf16 ===============================
__global__ __launch_bounds__(256) void av_kernel(
    const unsigned short* __restrict__ attn, const unsigned short* __restrict__ v,
    unsigned short* __restrict__ y)
{
    __shared__ alignas(16) unsigned short lA[2][128 * LDS_ST];
    __shared__ alignas(16) unsigned short lB[2][128 * LDS_ST];
    int b = blockIdx.y;
    int nb = blockIdx.x * 128;
    const unsigned short* ab = attn + (size_t)b * CR * CR;
    const unsigned short* vb = v + (size_t)b * CR * HW;
    unsigned short* yb = y + (size_t)b * CR * HW;

    int lane = threadIdx.x & 31, wave = threadIdx.x >> 5;
    int wm = (wave & 3) * 32, wn = (wave >> 2) * 64;
    v8f acc[2][4] = {};

    R2u4 ra = loadA_b16(ab, CR, 0);
    R2u4 rb = loadB_Tb16(vb, HW, 0, nb);
    storeA_b16(lA[0], ra);
    storeB_Tb16(lB[0], rb);
    __syncthreads();

    const int NCH = CR / KC;                   // 4
    for (int c = 0; c < NCH; ++c) {
        int cur = c & 1;
        if (c + 1 < NCH) {
            ra = loadA_b16(ab, CR, (c + 1) * KC);
            rb = loadB_Tb16(vb, HW, (c + 1) * KC, nb);
        }
        wmma_chunk(lA[cur], lB[cur], wm, wn, lane, acc);
        __syncthreads();
        if (c + 1 < NCH) {
            storeA_b16(lA[cur ^ 1], ra);
            storeB_Tb16(lB[cur ^ 1], rb);
        }
        __syncthreads();
    }

    int half = lane >> 4, lc = lane & 15;
#pragma unroll
    for (int i = 0; i < 2; ++i)
#pragma unroll
        for (int j = 0; j < 4; ++j)
#pragma unroll
            for (int r = 0; r < 8; ++r) {
                int m = wm + 16 * i + r + 8 * half;
                int n = nb + wn + 16 * j + lc;
                yb[(size_t)m * HW + n] = f32_to_bf16(acc[i][j][r]);
            }
}

// =============== kernel 5: out = prelu(o_w . y + o_b) -> fp32 ==================
__global__ __launch_bounds__(256) void out_proj_kernel(
    const unsigned short* __restrict__ y, const float* __restrict__ oW,
    const float* __restrict__ ob, const float* __restrict__ slope,
    float* __restrict__ out)
{
    __shared__ alignas(16) unsigned short lA[2][128 * LDS_ST];
    __shared__ alignas(16) unsigned short lB[2][128 * LDS_ST];
    int b  = blockIdx.z;
    int mb = blockIdx.y * 128;                 // 4 M tiles cover C=512
    int nb = blockIdx.x * 128;
    const unsigned short* yb = y + (size_t)b * CR * HW;
    const float* wb = oW + (size_t)mb * CR;

    int lane = threadIdx.x & 31, wave = threadIdx.x >> 5;
    int wm = (wave & 3) * 32, wn = (wave >> 2) * 64;
    v8f acc[2][4] = {};

    R4f4 ra = loadA_f32(wb, CR, 0);
    R2u4 rb = loadB_Tb16(yb, HW, 0, nb);
    storeA_f32(lA[0], ra);
    storeB_Tb16(lB[0], rb);
    __syncthreads();

    const int NCH = CR / KC;                   // 4
    for (int c = 0; c < NCH; ++c) {
        int cur = c & 1;
        if (c + 1 < NCH) {
            ra = loadA_f32(wb, CR, (c + 1) * KC);
            rb = loadB_Tb16(yb, HW, (c + 1) * KC, nb);
        }
        wmma_chunk(lA[cur], lB[cur], wm, wn, lane, acc);
        __syncthreads();
        if (c + 1 < NCH) {
            storeA_f32(lA[cur ^ 1], ra);
            storeB_Tb16(lB[cur ^ 1], rb);
        }
        __syncthreads();
    }

    float a = slope[0];
    int half = lane >> 4, lc = lane & 15;
#pragma unroll
    for (int i = 0; i < 2; ++i)
#pragma unroll
        for (int j = 0; j < 4; ++j)
#pragma unroll
            for (int r = 0; r < 8; ++r) {
                int m = mb + wm + 16 * i + r + 8 * half;
                int n = nb + wn + 16 * j + lc;
                float v = acc[i][j][r] + ob[m];
                v = v >= 0.f ? v : a * v;
                out[((size_t)b * CIN + m) * HW + n] = v;
            }
}

// ================================ launcher =====================================
extern "C" void kernel_launch(void* const* d_in, const int* in_sizes, int n_in,
                              void* d_out, int out_size, void* d_ws, size_t ws_size,
                              hipStream_t stream)
{
    (void)in_sizes; (void)n_in; (void)out_size; (void)ws_size;
    const float* x      = (const float*)d_in[0];
    const float* q_w    = (const float*)d_in[1];
    const float* q_b    = (const float*)d_in[2];
    const float* q_g    = (const float*)d_in[3];
    const float* q_beta = (const float*)d_in[4];
    const float* q_mu   = (const float*)d_in[5];
    const float* q_var  = (const float*)d_in[6];
    const float* q_a    = (const float*)d_in[7];
    const float* k_w    = (const float*)d_in[8];
    const float* k_b    = (const float*)d_in[9];
    const float* k_g    = (const float*)d_in[10];
    const float* k_beta = (const float*)d_in[11];
    const float* k_mu   = (const float*)d_in[12];
    const float* k_var  = (const float*)d_in[13];
    const float* k_a    = (const float*)d_in[14];
    const float* v_w    = (const float*)d_in[15];
    const float* v_b    = (const float*)d_in[16];
    const float* v_a    = (const float*)d_in[17];
    const float* o_w    = (const float*)d_in[18];
    const float* o_b    = (const float*)d_in[19];
    const float* o_a    = (const float*)d_in[20];

    unsigned char* ws = (unsigned char*)d_ws;
    const size_t SZP = (size_t)NB * CR * HW * sizeof(unsigned short);  // 37.75 MB
    unsigned short* wq   = (unsigned short*)(ws);
    unsigned short* wk   = (unsigned short*)(ws + SZP);
    unsigned short* wv   = (unsigned short*)(ws + 2 * SZP);
    unsigned short* wy   = (unsigned short*)(ws + 3 * SZP);
    float*          alp  = (float*)(ws + 4 * SZP);
    unsigned short* attn = (unsigned short*)(ws + 4 * SZP + (size_t)NB * CR * CR * sizeof(float));

    dim3 blk(256);
    dim3 gproj(HW / 128, NB);                  // 72 x 16
    qkv_proj_kernel<<<gproj, blk, 0, stream>>>(x, q_w, q_b, q_g, q_beta, q_mu, q_var, q_a, wq, 1);
    qkv_proj_kernel<<<gproj, blk, 0, stream>>>(x, k_w, k_b, k_g, k_beta, k_mu, k_var, k_a, wk, 1);
    qkv_proj_kernel<<<gproj, blk, 0, stream>>>(x, v_w, v_b, nullptr, nullptr, nullptr, nullptr, v_a, wv, 0);

    hipMemsetAsync(alp, 0, (size_t)NB * CR * CR * sizeof(float), stream);
    alpha_kernel<<<dim3(HW / 256, NB), blk, 0, stream>>>(wq, wk, alp);      // 36 x 16
    softmax_kernel<<<NB * CR, 128, 0, stream>>>(alp, attn);                 // 2048 rows
    av_kernel<<<dim3(HW / 128, NB), blk, 0, stream>>>(attn, wv, wy);        // 72 x 16
    out_proj_kernel<<<dim3(HW / 128, CIN / 128, NB), blk, 0, stream>>>(wy, o_w, o_b, o_a, (float*)d_out);
}
